// LocalGNOBlock_10539849744710
// MI455X (gfx1250) — matrix operand order repaired and checked by reading, not compiled, not run on hardware
//
#include <hip/hip_runtime.h>

#define NN 16384
#define BB 8
#define HH 128
#define KK 6
#define EPSV 1e-5f

typedef __attribute__((ext_vector_type(16))) __bf16 v16bf;
typedef __attribute__((ext_vector_type(8)))  float  v8f;
typedef __attribute__((ext_vector_type(4)))  unsigned int v4u;
typedef __attribute__((ext_vector_type(8)))  int v8i;
typedef __attribute__((ext_vector_type(4)))  int v4i;

union Frag {
  v16bf v;
  unsigned int   i[8];
  unsigned short u[16];
};

__device__ __forceinline__ float bf2f(unsigned short s) {
  return __uint_as_float(((unsigned int)s) << 16);
}

// Packed f32x2 -> bf16x2 convert (round-to-nearest-even).
#if __has_builtin(__builtin_amdgcn_cvt_pk_bf16_f32)
typedef __attribute__((ext_vector_type(2))) __bf16 v2bf;
__device__ __forceinline__ unsigned int pk2bf(float a, float b) {
  union { v2bf v; unsigned int u; } cv;
  cv.v = __builtin_amdgcn_cvt_pk_bf16_f32(a, b);
  return cv.u;
}
#else
__device__ __forceinline__ unsigned int pk2bf(float a, float b) {
  unsigned int r;
  asm("v_cvt_pk_bf16_f32 %0, %1, %2" : "=v"(r) : "v"(a), "v"(b));
  return r;
}
#endif
// Single f32 -> bf16 (in low 16 bits; b16 stores use VDATA[15:0]).
__device__ __forceinline__ unsigned short f2bf1(float x) {
  return (unsigned short)pk2bf(x, 0.0f);
}

__device__ __forceinline__ float silu(float x) {
  return x * (1.0f / (1.0f + __expf(-x)));
}
__device__ __forceinline__ v8f wmma_bf16(v16bf a, v16bf b, v8f c) {
  return __builtin_amdgcn_wmma_f32_16x16x32_bf16(false, a, false, b, (short)0, c,
                                                 false, false);
}
__device__ __forceinline__ int iclamp(int x, int lo, int hi) {
  return x < lo ? lo : (x > hi ? hi : x);
}

// ---------------------------------------------------------------------------
// Tensor Data Mover: 1-D contiguous global(bf16) -> LDS copy of nelem elements.
// ---------------------------------------------------------------------------
__device__ __forceinline__ unsigned ldsoff(const void* p) {
  return (unsigned)(uintptr_t)p;            // flat LDS aperture: offset in [31:0]
}
__device__ __forceinline__ void tdm_load_1d(unsigned lds_addr, const void* gsrc,
                                            unsigned nelem) {
  unsigned long long ga = (unsigned long long)(uintptr_t)gsrc;
  v4u g0;
  g0[0] = 1u;                                              // count=1
  g0[1] = lds_addr;                                        // LDS byte address
  g0[2] = (unsigned)(ga & 0xffffffffu);                    // global_addr[31:0]
  g0[3] = (unsigned)((ga >> 32) & 0x01ffffffu) | (2u << 30); // addr[56:32]|type=2
  v8i g1;
  g1[0] = (int)(1u << 16);                                 // data_size=1 (2 bytes)
  g1[1] = (int)((nelem & 0xffffu) << 16);                  // tensor_dim0 lo
  g1[2] = (int)((nelem >> 16) & 0xffffu) | (1 << 16);      // td0 hi | tensor_dim1=1
  g1[3] = (int)((nelem & 0xffffu) << 16);                  // tile_dim0
  g1[4] = 1;                                               // tile_dim1=1, tile_dim2=0
  g1[5] = (int)nelem;                                      // tensor_dim0_stride lo
  g1[6] = 0;
  g1[7] = 0;
  v4i z = {0, 0, 0, 0};
#if defined(__clang_major__) && __clang_major__ >= 23
  v8i z8 = {0, 0, 0, 0, 0, 0, 0, 0};
  __builtin_amdgcn_tensor_load_to_lds(g0, g1, z, z, z8, 0);
#else
  __builtin_amdgcn_tensor_load_to_lds(g0, g1, z, z, 0);
#endif
}
__device__ __forceinline__ void tdm_wait() {
  __builtin_amdgcn_s_wait_tensorcnt(0);
  asm volatile("" ::: "memory");
}

// Stage a 128x128 f32 weight (row-major [k][n]) into LDS pre-swizzled so that
// B-matrix fragments are contiguous 32B per lane: sW[ntile][kchunk][lane][8 dw].
__device__ __forceinline__ void stage_weight(const float* __restrict__ gw,
                                             unsigned int (*sW)[4][32][8],
                                             int tid, int nthr) {
  for (int idx = tid; idx < 8 * 4 * 32; idx += nthr) {
    int nt = idx >> 7;
    int kc = (idx >> 5) & 3;
    int L  = idx & 31;
    int kb = ((L & 16) ? 8 : 0) + 32 * kc;
    int n  = nt * 16 + (L & 15);
    unsigned int* dst = sW[nt][kc][L];
#pragma unroll
    for (int p = 0; p < 8; ++p)
      dst[p] = pk2bf(gw[(kb + 2 * p + (p >= 4 ? 8 : 0)) * HH + n], gw[(kb + 2 * p + (p >= 4 ? 8 : 0) + 1) * HH + n]);
  }
}

// A-matrix fragment (16x32 bf16) from a row-major 16x128 bf16 LDS tile.
__device__ __forceinline__ v16bf load_afrag(const unsigned short* X, int lane, int kc) {
  int r  = lane & 15;
  int kb = ((lane & 16) ? 8 : 0) + 32 * kc;
  const unsigned short* row = X + r * HH + kb;
  Frag f;
#pragma unroll
  for (int p = 0; p < 4; ++p) {
    f.i[p]     = *(const unsigned int*)(row + 2 * p);
    f.i[4 + p] = *(const unsigned int*)(row + 16 + 2 * p);
  }
  return f.v;
}
__device__ __forceinline__ v16bf load_bfrag(const unsigned int* w8) {
  Frag f;
#pragma unroll
  for (int p = 0; p < 8; ++p) f.i[p] = w8[p];
  return f.v;
}

// ---------------------------------------------------------------------------
// Kernel 1: A = h @ msg_w1[0:128] + b1 ; Bm = h @ msg_w1[128:256]   (bf16 out)
// ---------------------------------------------------------------------------
__global__ __launch_bounds__(64, 1)
void k_pre(const float* __restrict__ h, const float* __restrict__ msg_w1,
           const float* __restrict__ msg_b1,
           unsigned short* __restrict__ Aout, unsigned short* __restrict__ Bout) {
  __shared__ __align__(16) unsigned short sX[2][16][HH];     // 8 KB
  __shared__ __align__(16) unsigned int   sW[8][4][32][8];   // 32 KB
  int tid = threadIdx.x, w = tid >> 5, lane = tid & 31;
  long base = (long)blockIdx.x * 32 + w * 16;                // flat row over B*N

  __builtin_prefetch((const void*)(h + (base + 16) * HH), 0, 0);
  for (int r = 0; r < 16; ++r) {
    const float4 vv = *(const float4*)(h + (base + r) * HH + lane * 4);
    *(uint2*)&sX[w][r][lane * 4] = make_uint2(pk2bf(vv.x, vv.y), pk2bf(vv.z, vv.w));
  }
  for (int wsel = 0; wsel < 2; ++wsel) {
    __syncthreads();
    stage_weight(msg_w1 + (size_t)wsel * HH * HH, sW, tid, 64);
    __syncthreads();
    unsigned short* outp = wsel ? Bout : Aout;
    int r0 = (lane & 16) ? 8 : 0;
#pragma unroll
    for (int nt = 0; nt < 8; ++nt) {
      v8f acc = {};
#pragma unroll
      for (int kc = 0; kc < 4; ++kc)
        acc = wmma_bf16(load_afrag(&sX[w][0][0], lane, kc),
                        load_bfrag(sW[nt][kc][lane]), acc);
      int col = nt * 16 + (lane & 15);
      float bias = wsel ? 0.0f : msg_b1[col];
#pragma unroll
      for (int v = 0; v < 8; ++v)
        outp[(base + r0 + v) * HH + col] = f2bf1(acc[v] + bias);
    }
  }
}

// ---------------------------------------------------------------------------
// Kernel 2: agg[i] = ( sum_edges silu(A[i]+Bm[j]+dc*w1c) @ W2 + cnt*b2 ) / max(cnt,1)
// ---------------------------------------------------------------------------
__global__ __launch_bounds__(64, 1)
void k_msg(const unsigned short* __restrict__ Ain,
           const unsigned short* __restrict__ Bin,
           const float* __restrict__ coord, const float* __restrict__ msg_w1,
           const float* __restrict__ msg_w2, const float* __restrict__ msg_b2,
           unsigned short* __restrict__ agg) {
  __shared__ __align__(16) unsigned int   sW2[8][4][32][8];  // 32 KB
  __shared__ __align__(16) unsigned short sA[2][16][HH];     // 8 KB
  __shared__ __align__(16) unsigned short sB[2][28][HH];     // 14 KB
  __shared__ __align__(16) unsigned short sH[2][16][HH];     // 8 KB
  __shared__ float sC[2][28];
  int tid = threadIdx.x, w = tid >> 5, lane = tid & 31;
  int b  = blockIdx.x >> 9;                    // 512 tiles per batch (N/32)
  int tb = (blockIdx.x & 511) * 32 + w * 16;   // node index in batch
  long fb = (long)b * NN + tb;                 // flat tile-start row

  // A tile: contiguous 16*128 bf16 region -> TDM async copy to LDS.
  tdm_load_1d(ldsoff(&sA[w][0][0]), Ain + fb * HH, 16 * HH);

  // B window rows tb-6 .. tb+21: TDM when fully interior, else clamped copy.
  bool interior = (tb >= 6) && (tb + 21 < NN);
  if (interior) {
    tdm_load_1d(ldsoff(&sB[w][0][0]), Bin + (fb - 6) * HH, 28 * HH);
  } else {
    for (int r = 0; r < 28; ++r) {
      int j = iclamp(tb - 6 + r, 0, NN - 1);
      *(uint2*)&sB[w][r][lane * 4] =
          *(const uint2*)(Bin + ((long)b * NN + j) * HH + lane * 4);
    }
  }
  stage_weight(msg_w2, sW2, tid, 64);
  if (lane < 28) sC[w][lane] = coord[(long)b * NN + iclamp(tb - 6 + lane, 0, NN - 1)];
  float w1c[4];
#pragma unroll
  for (int q = 0; q < 4; ++q) w1c[q] = msg_w1[(size_t)256 * HH + lane * 4 + q];

  v8f zero = {};
  v8f acc[8];
#pragma unroll
  for (int nt = 0; nt < 8; ++nt) acc[nt] = zero;
  tdm_wait();                    // sA/sB resident before use
  __syncthreads();               // sW2 staged by both waves

  for (int e = 0; e < 12; ++e) {
    int d = (e < 6) ? (e + 1) : (5 - e);   // 1..6 then -1..-6
    for (int r = 0; r < 16; ++r) {
      int i = tb + r, j = i + d;
      bool valid = (j >= 0) && (j < NN);
      int wr = r + d + 6;                  // window row in [0,27]
      float dc = sC[w][wr] - sC[w][r + 6];
      int c = lane * 4;
      float h0 = silu(bf2f(sA[w][r][c + 0]) + bf2f(sB[w][wr][c + 0]) + dc * w1c[0]);
      float h1 = silu(bf2f(sA[w][r][c + 1]) + bf2f(sB[w][wr][c + 1]) + dc * w1c[1]);
      float h2 = silu(bf2f(sA[w][r][c + 2]) + bf2f(sB[w][wr][c + 2]) + dc * w1c[2]);
      float h3 = silu(bf2f(sA[w][r][c + 3]) + bf2f(sB[w][wr][c + 3]) + dc * w1c[3]);
      if (!valid) { h0 = h1 = h2 = h3 = 0.0f; }
      *(uint2*)&sH[w][r][c] = make_uint2(pk2bf(h0, h1), pk2bf(h2, h3));
    }
    __syncthreads();
    v16bf hf[4];
#pragma unroll
    for (int kc = 0; kc < 4; ++kc) hf[kc] = load_afrag(&sH[w][0][0], lane, kc);
#pragma unroll
    for (int nt = 0; nt < 8; ++nt)
#pragma unroll
      for (int kc = 0; kc < 4; ++kc)
        acc[nt] = wmma_bf16(hf[kc], load_bfrag(sW2[nt][kc][lane]), acc[nt]);
    __syncthreads();
  }

  int r0 = (lane & 16) ? 8 : 0;
#pragma unroll
  for (int nt = 0; nt < 8; ++nt) {
    int col = nt * 16 + (lane & 15);
    float b2 = msg_b2[col];
#pragma unroll
    for (int v = 0; v < 8; ++v) {
      int i = tb + r0 + v;
      int lo = i < KK ? i : KK;
      int hi = (NN - 1 - i) < KK ? (NN - 1 - i) : KK;
      float cnt = (float)(lo + hi);
      float val = (acc[nt][v] + cnt * b2) / fmaxf(cnt, 1.0f);
      agg[(fb + r0 + v) * HH + col] = f2bf1(val);
    }
  }
}

// ---------------------------------------------------------------------------
// Kernel 3: x = h + silu(h@U1a + agg@U1b + b1)@U2 + b2 ; out = LayerNorm(x)
// ---------------------------------------------------------------------------
__global__ __launch_bounds__(64, 1)
void k_upd(const float* __restrict__ h, const unsigned short* __restrict__ agg,
           const float* __restrict__ upd_w1, const float* __restrict__ upd_b1,
           const float* __restrict__ upd_w2, const float* __restrict__ upd_b2,
           const float* __restrict__ ln_g, const float* __restrict__ ln_b,
           float* __restrict__ out) {
  __shared__ __align__(16) unsigned int   sW[8][4][32][8];   // 32 KB (reused 3x)
  __shared__ __align__(16) unsigned short sXh[2][16][HH];    // 8 KB
  __shared__ __align__(16) unsigned short sXa[2][16][HH];    // 8 KB
  __shared__ __align__(16) unsigned short sH[2][16][HH];     // 8 KB
  int tid = threadIdx.x, w = tid >> 5, lane = tid & 31;
  long base = (long)blockIdx.x * 32 + w * 16;

  // agg tile is a contiguous bf16 region -> TDM async copy to LDS.
  tdm_load_1d(ldsoff(&sXa[w][0][0]), agg + base * HH, 16 * HH);

  __builtin_prefetch((const void*)(h + base * HH), 0, 0);
  for (int r = 0; r < 16; ++r) {
    const float4 vv = *(const float4*)(h + (base + r) * HH + lane * 4);
    *(uint2*)&sXh[w][r][lane * 4] = make_uint2(pk2bf(vv.x, vv.y), pk2bf(vv.z, vv.w));
  }
  v8f zero = {};
  v8f acc[8];
#pragma unroll
  for (int nt = 0; nt < 8; ++nt) acc[nt] = zero;

  __syncthreads();
  stage_weight(upd_w1, sW, tid, 64);            // U1a (rows 0..127)
  __syncthreads();
#pragma unroll
  for (int nt = 0; nt < 8; ++nt)
#pragma unroll
    for (int kc = 0; kc < 4; ++kc)
      acc[nt] = wmma_bf16(load_afrag(&sXh[w][0][0], lane, kc),
                          load_bfrag(sW[nt][kc][lane]), acc[nt]);
  __syncthreads();
  stage_weight(upd_w1 + (size_t)HH * HH, sW, tid, 64);   // U1b (rows 128..255)
  __syncthreads();
  tdm_wait();                                   // sXa resident
#pragma unroll
  for (int nt = 0; nt < 8; ++nt)
#pragma unroll
    for (int kc = 0; kc < 4; ++kc)
      acc[nt] = wmma_bf16(load_afrag(&sXa[w][0][0], lane, kc),
                          load_bfrag(sW[nt][kc][lane]), acc[nt]);

  // SiLU(acc + b1) -> hidden tile in LDS (C/D layout -> row major)
  int r0 = (lane & 16) ? 8 : 0;
#pragma unroll
  for (int nt = 0; nt < 8; ++nt) {
    int col = nt * 16 + (lane & 15);
    float b1 = upd_b1[col];
#pragma unroll
    for (int v = 0; v < 8; ++v)
      sH[w][r0 + v][col] = f2bf1(silu(acc[nt][v] + b1));
  }
  __syncthreads();
  stage_weight(upd_w2, sW, tid, 64);
  __syncthreads();
#pragma unroll
  for (int nt = 0; nt < 8; ++nt) acc[nt] = zero;
#pragma unroll
  for (int nt = 0; nt < 8; ++nt)
#pragma unroll
    for (int kc = 0; kc < 4; ++kc)
      acc[nt] = wmma_bf16(load_afrag(&sH[w][0][0], lane, kc),
                          load_bfrag(sW[nt][kc][lane]), acc[nt]);

  // residual + per-row LayerNorm (rows split across lane half-groups)
  float lg[8], lb[8], b2v[8];
#pragma unroll
  for (int nt = 0; nt < 8; ++nt) {
    int col = nt * 16 + (lane & 15);
    lg[nt] = ln_g[col]; lb[nt] = ln_b[col]; b2v[nt] = upd_b2[col];
  }
#pragma unroll
  for (int nt = 0; nt < 8; ++nt) {
    int col = nt * 16 + (lane & 15);
#pragma unroll
    for (int v = 0; v < 8; ++v)
      acc[nt][v] = acc[nt][v] + b2v[nt] + h[(base + r0 + v) * HH + col];
  }
#pragma unroll
  for (int v = 0; v < 8; ++v) {
    float s = 0.0f, ss = 0.0f;
#pragma unroll
    for (int nt = 0; nt < 8; ++nt) { float x = acc[nt][v]; s += x; ss += x * x; }
    for (int m = 1; m <= 8; m <<= 1) {
      s  += __shfl_xor(s,  m, 32);
      ss += __shfl_xor(ss, m, 32);
    }
    float mean = s * (1.0f / HH);
    float var  = ss * (1.0f / HH) - mean * mean;
    float rstd = rsqrtf(var + EPSV);
#pragma unroll
    for (int nt = 0; nt < 8; ++nt) {
      int col = nt * 16 + (lane & 15);
      out[(base + r0 + v) * HH + col] = (acc[nt][v] - mean) * rstd * lg[nt] + lb[nt];
    }
  }
}

extern "C" void kernel_launch(void* const* d_in, const int* in_sizes, int n_in,
                              void* d_out, int out_size, void* d_ws, size_t ws_size,
                              hipStream_t stream) {
  (void)in_sizes; (void)n_in; (void)out_size; (void)ws_size;
  const float* h      = (const float*)d_in[0];
  const float* coord  = (const float*)d_in[1];
  const float* msg_w1 = (const float*)d_in[2];
  const float* msg_b1 = (const float*)d_in[3];
  const float* msg_w2 = (const float*)d_in[4];
  const float* msg_b2 = (const float*)d_in[5];
  const float* upd_w1 = (const float*)d_in[6];
  const float* upd_b1 = (const float*)d_in[7];
  const float* upd_w2 = (const float*)d_in[8];
  const float* upd_b2 = (const float*)d_in[9];
  const float* ln_g   = (const float*)d_in[10];
  const float* ln_b   = (const float*)d_in[11];
  float* out = (float*)d_out;

  size_t plane = (size_t)BB * NN * HH * sizeof(unsigned short);   // 32 MB
  unsigned short* A   = (unsigned short*)d_ws;
  unsigned short* Bm  = (unsigned short*)((char*)d_ws + plane);
  unsigned short* agp = (unsigned short*)((char*)d_ws + 2 * plane);

  dim3 grid(BB * NN / 32), block(64);
  k_pre<<<grid, block, 0, stream>>>(h, msg_w1, msg_b1, A, Bm);
  k_msg<<<grid, block, 0, stream>>>(A, Bm, coord, msg_w1, msg_w2, msg_b2, agp);
  k_upd<<<grid, block, 0, stream>>>(h, agp, upd_w1, upd_b1, upd_w2, upd_b2,
                                    ln_g, ln_b, out);
}
